// MoEFFN_25744033972352
// MI455X (gfx1250) — matrix operand order repaired and checked by reading, not compiled, not run on hardware
//
#include <hip/hip_runtime.h>
#include <hip/hip_bf16.h>
#include <cstdint>

#define T_TOK   4096
#define DM      1024
#define H_DIM   4096
#define NE      8
#define CH      256      // H chunk per outer iteration
#define NPW     128      // output columns per wave
#define XP      1032     // padded Xs row stride (bf16): 1032*2B, +4 dwords -> conflict-free
#define HP      264      // padded Hs row stride (bf16)

typedef __attribute__((ext_vector_type(16))) __bf16 v16bf;
typedef __attribute__((ext_vector_type(8)))  float  v8f;

// gelu_tanh(v) = v * sigmoid(2*sqrt(2/pi)*(v + 0.044715 v^3)); branchless.
__device__ __forceinline__ float gelu_tanh(float v) {
    float u = v + 0.044715f * v * v * v;
    // exp(-1.59576912*u * ... ) via exp2: 2*sqrt(2/pi)/ln2 = 2.302204213
    float z = __builtin_amdgcn_exp2f(-2.302204213f * u);
    return v * __builtin_amdgcn_rcpf(1.0f + z);
}

// WMMA A-operand (16x32 bf16) from a row-major bf16 LDS tile.
__device__ __forceinline__ v16bf load_a_lds(const __bf16* rowptr, int kc, int lane) {
    int hi = lane >> 4;
    const __bf16* p0 = rowptr + kc + hi * 16;
    const __bf16* p1 = rowptr + kc + hi * 16 + 8;
    v16bf a;
#pragma unroll
    for (int i = 0; i < 8; ++i) { a[i] = p0[i]; a[8 + i] = p1[i]; }
    return a;
}

// 16 contiguous bf16 (32 B) from global -> B-operand half-row.
__device__ __forceinline__ v16bf load16bf(const __bf16* p) {
    v16bf b;
#pragma unroll
    for (int i = 0; i < 16; ++i) b[i] = p[i];
    return b;
}

// 16 contiguous fp32 from global, convert (fallback path).
__device__ __forceinline__ v16bf cvt16(const float* p) {
    const float4* q = (const float4*)p;
    v16bf b;
#pragma unroll
    for (int i = 0; i < 4; ++i) {
        float4 t = q[i];
        b[4 * i + 0] = (__bf16)t.x;  b[4 * i + 1] = (__bf16)t.y;
        b[4 * i + 2] = (__bf16)t.z;  b[4 * i + 3] = (__bf16)t.w;
    }
    return b;
}

__global__ void moe_init(float* out, int* counts) {
    int i = blockIdx.x * blockDim.x + threadIdx.x;
    if (i < T_TOK * DM) out[i] = 0.0f;
    if (i < NE) counts[i] = 0;
}

__global__ void cvt_f32_to_bf16(const float* __restrict__ src, __bf16* __restrict__ dst) {
    int i = (blockIdx.x * blockDim.x + threadIdx.x) * 4;
    float4 v = *(const float4*)(src + i);
    dst[i + 0] = (__bf16)v.x;  dst[i + 1] = (__bf16)v.y;
    dst[i + 2] = (__bf16)v.z;  dst[i + 3] = (__bf16)v.w;
}

__global__ __launch_bounds__(256)
void moe_router(const float* __restrict__ x, const float* __restrict__ rw,
                int* __restrict__ counts, int* __restrict__ tok_idx,
                float* __restrict__ tok_w) {
    __shared__ float srw[NE * DM];
    for (int i = threadIdx.x; i < NE * DM; i += blockDim.x) srw[i] = rw[i];
    __syncthreads();

    int t = blockIdx.x * blockDim.x + threadIdx.x;
    if (t >= T_TOK) return;
    const float* xt = x + (size_t)t * DM;

    float lg[NE];
#pragma unroll
    for (int e = 0; e < NE; ++e) lg[e] = 0.0f;
#pragma unroll 4
    for (int d = 0; d < DM; ++d) {
        float xv = xt[d];
#pragma unroll
        for (int e = 0; e < NE; ++e) lg[e] += xv * srw[e * DM + d];
    }

    int i0 = 0; float v0 = lg[0];
#pragma unroll
    for (int e = 1; e < NE; ++e) if (lg[e] > v0) { v0 = lg[e]; i0 = e; }
    int i1 = -1; float v1 = -3.402823e38f;
#pragma unroll
    for (int e = 0; e < NE; ++e) if (e != i0 && lg[e] > v1) { v1 = lg[e]; i1 = e; }

    float m  = fmaxf(v0, v1);
    float e0 = expf(v0 - m), e1 = expf(v1 - m);
    float s  = e0 + e1;
    float w0 = e0 / s, w1 = e1 / s;

    int p0 = atomicAdd(&counts[i0], 1);
    tok_idx[i0 * T_TOK + p0] = t;  tok_w[i0 * T_TOK + p0] = w0;
    int p1 = atomicAdd(&counts[i1], 1);
    tok_idx[i1 * T_TOK + p1] = t;  tok_w[i1 * T_TOK + p1] = w1;
}

// ---------------- bf16-weight fast path: 32-token tiles ----------------
__global__ __launch_bounds__(256)
void moe_ffn_bf16(const __bf16* __restrict__ xbf,
                  const __bf16* __restrict__ w1bf, const float* __restrict__ fc1b,
                  const __bf16* __restrict__ w2bf, const float* __restrict__ fc2b,
                  const int* __restrict__ counts, const int* __restrict__ tok_idx,
                  const float* __restrict__ tok_w, float* __restrict__ out) {
    int e    = blockIdx.y;
    int tile = blockIdx.x;
    int cnt  = counts[e];
    int ntiles = (cnt + 31) >> 5;
    if (tile >= ntiles) return;               // uniform across block

    __shared__ __bf16 Xs[32][XP];             // ~64.5 KB gathered token tile
    __shared__ __bf16 Hs[32][HP];             // ~16.5 KB GELU(h) chunk
    __shared__ int    sidx[32];
    __shared__ float  swt[32];

    int tid  = threadIdx.x;
    int lane = tid & 31;
    int wave = tid >> 5;

    if (tid < 32) {
        int p = tile * 32 + tid;
        sidx[tid] = (p < cnt) ? tok_idx[e * T_TOK + p] : 0;
        swt[tid]  = (p < cnt) ? tok_w[e * T_TOK + p] : 0.0f;   // pad rows -> weight 0
    }
    __syncthreads();

    // Async gather: 32 rows x 2048 B; 8 threads/row, 256 B each (16 x b128),
    // straight into LDS via the CDNA5 async-to-LDS path (ASYNCcnt).
    {
        int row = tid >> 3;
        int sub = tid & 7;
        const char* gp = (const char*)(xbf + (size_t)sidx[row] * DM) + sub * 256;
        unsigned    lo = (unsigned)(uintptr_t)((char*)&Xs[row][0] + sub * 256);
#pragma unroll
        for (int i = 0; i < 16; ++i) {
            unsigned    l = lo + i * 16;
            const void* g = gp + i * 16;
            asm volatile("global_load_async_to_lds_b128 %0, %1, off"
                         :: "v"(l), "v"(g) : "memory");
        }
        asm volatile("s_wait_asynccnt 0x0" ::: "memory");
    }
    __syncthreads();

    int col = lane & 15;
    int khi = (lane >> 4) * 16;
    int rhi = (lane >> 4) * 8;
    int nb  = wave * NPW;

    v8f acc[16];                              // [0..7]=rows 0-15, [8..15]=rows 16-31
#pragma unroll
    for (int j = 0; j < 16; ++j) acc[j] = (v8f){};

    const size_t w1e = (size_t)e * H_DIM * DM;
    const size_t w2e = (size_t)e * DM * H_DIM;
    const __bf16* xr0 = &Xs[lane & 15][0];
    const __bf16* xr1 = &Xs[16 + (lane & 15)][0];
    const __bf16* hr0 = &Hs[lane & 15][0];
    const __bf16* hr1 = &Hs[16 + (lane & 15)][0];
    const __bf16* w2base = w2bf + w2e + (size_t)(nb + col) * H_DIM;

    for (int hc = 0; hc < H_DIM; hc += CH) {
        // ---- FC1: wave's 32 h-columns x 32 rows; each B feeds two A row-tiles ----
        int hb = hc + wave * 32;
        const __bf16* w1r0 = w1bf + w1e + (size_t)(hb + col) * DM;
        const __bf16* w1r1 = w1r0 + (size_t)16 * DM;
        v8f c00 = (v8f){}, c01 = (v8f){}, c10 = (v8f){}, c11 = (v8f){};
        for (int kc = 0; kc < DM; kc += 32) {
            __builtin_prefetch(w1r0 + kc + khi + 256, 0, 1);
            v16bf a0 = load_a_lds(xr0, kc, lane);
            v16bf a1 = load_a_lds(xr1, kc, lane);
            v16bf b0 = load16bf(w1r0 + kc + khi);
            v16bf b1 = load16bf(w1r1 + kc + khi);
            c00 = __builtin_amdgcn_wmma_f32_16x16x32_bf16(false, a0, false, b0, (short)0, c00, false, false);
            c10 = __builtin_amdgcn_wmma_f32_16x16x32_bf16(false, a1, false, b0, (short)0, c10, false, false);
            c01 = __builtin_amdgcn_wmma_f32_16x16x32_bf16(false, a0, false, b1, (short)0, c01, false, false);
            c11 = __builtin_amdgcn_wmma_f32_16x16x32_bf16(false, a1, false, b1, (short)0, c11, false, false);
        }
        float bias0 = fc1b[e * H_DIM + hb + col];
        float bias1 = fc1b[e * H_DIM + hb + 16 + col];
#pragma unroll
        for (int r = 0; r < 8; ++r) {
            Hs[r + rhi][wave * 32 + col]           = (__bf16)gelu_tanh(c00[r] + bias0);
            Hs[r + rhi][wave * 32 + 16 + col]      = (__bf16)gelu_tanh(c01[r] + bias1);
            Hs[16 + r + rhi][wave * 32 + col]      = (__bf16)gelu_tanh(c10[r] + bias0);
            Hs[16 + r + rhi][wave * 32 + 16 + col] = (__bf16)gelu_tanh(c11[r] + bias1);
        }
        __syncthreads();

        // ---- FC2 partial: one B fragment feeds both row-tiles; A feeds 8 N-tiles ----
        for (int kk = 0; kk < CH; kk += 32) {
            v16bf a0 = load_a_lds(hr0, kk, lane);
            v16bf a1 = load_a_lds(hr1, kk, lane);
            const __bf16* wp = w2base + hc + kk + khi;
#pragma unroll
            for (int j = 0; j < 8; ++j) {
                v16bf b = load16bf(wp + (size_t)j * 16 * H_DIM);
                acc[j]     = __builtin_amdgcn_wmma_f32_16x16x32_bf16(false, a0, false, b, (short)0, acc[j], false, false);
                acc[8 + j] = __builtin_amdgcn_wmma_f32_16x16x32_bf16(false, a1, false, b, (short)0, acc[8 + j], false, false);
            }
        }
        __syncthreads();
    }

#pragma unroll
    for (int j = 0; j < 8; ++j) {
        int o = nb + j * 16 + col;
        float b2 = fc2b[e * DM + o];
#pragma unroll
        for (int r = 0; r < 8; ++r) {
            int r0 = r + rhi;
            int r1 = 16 + r + rhi;
            atomicAdd(&out[(size_t)sidx[r0] * DM + o], (acc[j][r]     + b2) * swt[r0]);
            atomicAdd(&out[(size_t)sidx[r1] * DM + o], (acc[8 + j][r] + b2) * swt[r1]);
        }
    }
}

// ---------------- fp32-weight fallback (small ws) ----------------
__global__ __launch_bounds__(256)
void moe_ffn_f32(const float* __restrict__ x,
                 const float* __restrict__ fc1w, const float* __restrict__ fc1b,
                 const float* __restrict__ fc2w, const float* __restrict__ fc2b,
                 const int* __restrict__ counts, const int* __restrict__ tok_idx,
                 const float* __restrict__ tok_w, float* __restrict__ out) {
    int e    = blockIdx.y;
    int tile = blockIdx.x;
    int cnt  = counts[e];
    int ntiles = (cnt + 15) >> 4;
    if (tile >= ntiles) return;

    __shared__ __bf16 Xs[16][XP];
    __shared__ __bf16 Hs[16][HP];
    __shared__ int    sidx[16];
    __shared__ float  swt[16];

    int tid  = threadIdx.x;
    int lane = tid & 31;
    int wave = tid >> 5;

    if (tid < 16) {
        int p = tile * 16 + tid;
        sidx[tid] = (p < cnt) ? tok_idx[e * T_TOK + p] : 0;
        swt[tid]  = (p < cnt) ? tok_w[e * T_TOK + p] : 0.0f;
    }
    __syncthreads();

    {
        int row = tid >> 4;
        int sub = tid & 15;
        const float4* src = (const float4*)(x + (size_t)sidx[row] * DM) + sub * 16;
#pragma unroll 4
        for (int i = 0; i < 16; ++i) {
            float4 v = src[i];
            int c = sub * 64 + i * 4;
            Xs[row][c + 0] = (__bf16)v.x;  Xs[row][c + 1] = (__bf16)v.y;
            Xs[row][c + 2] = (__bf16)v.z;  Xs[row][c + 3] = (__bf16)v.w;
        }
    }
    __syncthreads();

    int col = lane & 15;
    int khi = (lane >> 4) * 16;
    int rhi = (lane >> 4) * 8;
    int nb  = wave * NPW;

    v8f acc[8];
#pragma unroll
    for (int j = 0; j < 8; ++j) acc[j] = (v8f){};

    const size_t w1e = (size_t)e * H_DIM * DM;
    const size_t w2e = (size_t)e * DM * H_DIM;
    const __bf16* xrow = &Xs[lane & 15][0];
    const __bf16* hrow = &Hs[lane & 15][0];

    for (int hc = 0; hc < H_DIM; hc += CH) {
        int hb = hc + wave * 32;
        const float* w1r0 = fc1w + w1e + (size_t)(hb + col) * DM;
        const float* w1r1 = w1r0 + (size_t)16 * DM;
        v8f c0 = (v8f){}, c1 = (v8f){};
        for (int kc = 0; kc < DM; kc += 32) {
            v16bf a  = load_a_lds(xrow, kc, lane);
            v16bf b0 = cvt16(w1r0 + kc + khi);
            v16bf b1 = cvt16(w1r1 + kc + khi);
            c0 = __builtin_amdgcn_wmma_f32_16x16x32_bf16(false, a, false, b0, (short)0, c0, false, false);
            c1 = __builtin_amdgcn_wmma_f32_16x16x32_bf16(false, a, false, b1, (short)0, c1, false, false);
        }
        float bias0 = fc1b[e * H_DIM + hb + col];
        float bias1 = fc1b[e * H_DIM + hb + 16 + col];
#pragma unroll
        for (int r = 0; r < 8; ++r) {
            Hs[r + rhi][wave * 32 + col]      = (__bf16)gelu_tanh(c0[r] + bias0);
            Hs[r + rhi][wave * 32 + 16 + col] = (__bf16)gelu_tanh(c1[r] + bias1);
        }
        __syncthreads();

        const float* w2base = fc2w + w2e + (size_t)(nb + col) * H_DIM + hc;
        for (int kk = 0; kk < CH; kk += 32) {
            v16bf a = load_a_lds(hrow, kk, lane);
#pragma unroll
            for (int j = 0; j < 8; ++j) {
                v16bf b = cvt16(w2base + (size_t)j * 16 * H_DIM + kk + khi);
                acc[j] = __builtin_amdgcn_wmma_f32_16x16x32_bf16(false, a, false, b, (short)0, acc[j], false, false);
            }
        }
        __syncthreads();
    }

#pragma unroll
    for (int j = 0; j < 8; ++j) {
        int o = nb + j * 16 + col;
        float b2 = fc2b[e * DM + o];
#pragma unroll
        for (int r = 0; r < 8; ++r) {
            int row = r + rhi;
            atomicAdd(&out[(size_t)sidx[row] * DM + o], (acc[j][r] + b2) * swt[row]);
        }
    }
}

extern "C" void kernel_launch(void* const* d_in, const int* in_sizes, int n_in,
                              void* d_out, int out_size, void* d_ws, size_t ws_size,
                              hipStream_t stream) {
    const float* x   = (const float*)d_in[0];
    const float* rw  = (const float*)d_in[1];
    const float* w1  = (const float*)d_in[2];
    const float* b1  = (const float*)d_in[3];
    const float* w2  = (const float*)d_in[4];
    const float* b2  = (const float*)d_in[5];
    float* out = (float*)d_out;

    char* ws = (char*)d_ws;
    int*   counts  = (int*)ws;                                    // 64 B
    int*   tok_idx = (int*)(ws + 64);                             // 128 KB
    float* tok_w   = (float*)(ws + 64 + NE * T_TOK * 4);          // 128 KB
    size_t base    = 64 + (size_t)NE * T_TOK * 8;

    const size_t nx  = (size_t)T_TOK * DM;                        // 4 M elems
    const size_t nw  = (size_t)NE * H_DIM * DM;                   // 33.5 M elems
    __bf16* xbf  = (__bf16*)(ws + base);
    __bf16* w1bf = (__bf16*)(ws + base + nx * 2);
    __bf16* w2bf = (__bf16*)(ws + base + nx * 2 + nw * 2);
    size_t need  = base + (nx + 2 * nw) * 2;

    moe_init<<<(T_TOK * DM + 255) / 256, 256, 0, stream>>>(out, counts);
    moe_router<<<T_TOK / 256, 256, 0, stream>>>(x, rw, counts, tok_idx, tok_w);

    if (ws_size >= need) {
        cvt_f32_to_bf16<<<nx / (256 * 4), 256, 0, stream>>>(x, xbf);
        cvt_f32_to_bf16<<<nw / (256 * 4), 256, 0, stream>>>(w1, w1bf);
        cvt_f32_to_bf16<<<nw / (256 * 4), 256, 0, stream>>>(w2, w2bf);
        dim3 grid(T_TOK / 32, NE);
        moe_ffn_bf16<<<grid, 256, 0, stream>>>(xbf, w1bf, b1, w2bf, b2,
                                               counts, tok_idx, tok_w, out);
    } else {
        dim3 grid(T_TOK / 16, NE);
        moe_ffn_f32<<<grid, 256, 0, stream>>>(x, w1, b1, w2, b2,
                                              counts, tok_idx, tok_w, out);
    }
}
